// LeNet5_3066606649422
// MI455X (gfx1250) — compile-verified, matrix operand order checked by reading, not run on hardware
//
#include <hip/hip_runtime.h>
#include <hip/hip_bf16.h>
#include <math.h>

typedef __attribute__((ext_vector_type(16))) _Float16 v16h;
typedef __attribute__((ext_vector_type(8)))  _Float16 v8h;
typedef __attribute__((ext_vector_type(4)))  _Float16 v4h;
typedef __attribute__((ext_vector_type(8)))  float    v8f;

#define NB 8192

__device__ __forceinline__ float squash(float v) {
    return 1.7159f * tanhf(0.66666667f * v);
}

// Combine two 8-halfword chunks (chunk1: elems 0-7 = K kc+half*8..+7,
// chunk2: elems 8-15 = K kc+16+half*8..+7) into one WMMA f16 fragment.
__device__ __forceinline__ v16h frag_cat(v8h lo, v8h hi) {
    return __builtin_shufflevector(lo, hi, 0,1,2,3,4,5,6,7,8,9,10,11,12,13,14,15);
}

// LeNet-5 S2->C3 connectivity (MAP_INFO), rows = input channel, cols = output channel
__constant__ int MAPC[6][16] = {
    {1,0,0,0,1,1,1,0,0,1,1,1,1,0,1,1},
    {1,1,0,0,0,1,1,1,0,0,1,1,1,1,0,1},
    {1,1,1,0,0,0,1,1,1,0,0,1,0,1,1,1},
    {0,1,1,1,0,0,1,1,1,1,0,0,1,0,1,1},
    {0,0,1,1,1,0,0,1,1,1,1,0,1,1,0,1},
    {0,0,0,1,1,1,0,0,1,1,1,1,0,1,1,1},
};

// ---------------------------------------------------------------------------
// K1: C1 (5x5 conv, replicate pad 2) + S2 (2x2 avg pool, affine) + tanh.
// One block per image; image staged in LDS via float4; each thread -> 6 outs.
// h2: f16 [B,6,16,16]
// ---------------------------------------------------------------------------
__global__ void k1_c1_s2(const float* __restrict__ x,
                         const float* __restrict__ c1w,
                         const float* __restrict__ c1b,
                         const float* __restrict__ s2w,
                         const float* __restrict__ s2b,
                         _Float16* __restrict__ h2) {
    __shared__ __align__(16) float img[1024];
    __shared__ float w[150];
    __shared__ float cb[6], pw[6], pb[6];
    int t = threadIdx.x;
    int b = blockIdx.x;
    ((float4*)img)[t] = ((const float4*)(x + (size_t)b * 1024))[t];
    if (t < 150) w[t] = c1w[t];
    if (t < 6) { cb[t] = c1b[t]; pw[t] = s2w[t]; pb[t] = s2b[t]; }
    __syncthreads();
    #pragma unroll
    for (int kk = 0; kk < 6; ++kk) {
        int idx = kk * 256 + t;              // c*256 + py*16 + px
        int c = idx >> 8, pos = idx & 255, py = pos >> 4, px = pos & 15;
        const float* wc = &w[c * 25];
        float s = 0.f;
        #pragma unroll
        for (int dy = 0; dy < 2; ++dy)
            #pragma unroll
            for (int dx = 0; dx < 2; ++dx) {
                int y = 2 * py + dy, x0 = 2 * px + dx;
                #pragma unroll
                for (int ky = 0; ky < 5; ++ky) {
                    int yy = y + ky - 2; yy = yy < 0 ? 0 : (yy > 31 ? 31 : yy);
                    #pragma unroll
                    for (int kx = 0; kx < 5; ++kx) {
                        int xx = x0 + kx - 2; xx = xx < 0 ? 0 : (xx > 31 ? 31 : xx);
                        s = fmaf(wc[ky * 5 + kx], img[yy * 32 + xx], s);
                    }
                }
            }
        float v = (s * 0.25f + cb[c]) * pw[c] + pb[c];
        h2[(size_t)b * 1536 + idx] = (_Float16)squash(v);
    }
}

// ---------------------------------------------------------------------------
// K2: pack f16 weight panels in [N][K] layout (lane column contiguous ->
// B fragments become 2x global_load_b128) and zero the K-pads of A5/A6.
//   Wc3T: [16][160], Wc5T: [128][416], Wf6T: [96][128]
// ---------------------------------------------------------------------------
__global__ void k2_pack(const float* __restrict__ c3w,
                        const float* __restrict__ c5w,
                        const float* __restrict__ f6w,
                        _Float16* __restrict__ Wc3T,
                        _Float16* __restrict__ Wc5T,
                        _Float16* __restrict__ Wf6T,
                        _Float16* __restrict__ A5,
                        _Float16* __restrict__ A6) {
    int i = blockIdx.x * 256 + threadIdx.x;
    if (i < 2560) {                                   // Wc3T
        int n = i / 160, k = i % 160;
        float v = 0.f;
        if (k < 150 && MAPC[k / 25][n]) v = c3w[n * 150 + k];
        Wc3T[i] = (_Float16)v;
    } else if (i < 2560 + 53248) {                    // Wc5T
        int j = i - 2560;
        int n = j / 416, k = j % 416;
        float v = (k < 400 && n < 120) ? c5w[n * 400 + k] : 0.f;
        Wc5T[j] = (_Float16)v;
    } else if (i < 2560 + 53248 + 12288) {            // Wf6T
        int j = i - (2560 + 53248);
        int n = j >> 7, k = j & 127;
        float v = (k < 120 && n < 84) ? f6w[n * 120 + k] : 0.f;
        Wf6T[j] = (_Float16)v;
    } else if (i < 2560 + 53248 + 12288 + 524288) {   // A5 pad cols [400,416)
        int j = i - (2560 + 53248 + 12288);
        A5[(size_t)(j >> 4) * 416 + 400 + (j & 15)] = (_Float16)0.f;
    } else {                                          // A6 pad cols [120,128)
        int j = i - (2560 + 53248 + 12288 + 524288);  // < 262144
        A6[(size_t)(j >> 3) * 128 + 120 + (j & 7)] = (_Float16)0.f;
    }
}

// ---------------------------------------------------------------------------
// K3: C3 as WMMA GEMM.  One 288-thread block (9 waves) per image.
// M = 144 pixels (9 tiles of 16), N = 16 channels, K = 160.
// Image h2 staged in LDS (+ zero pad); k->LDS-offset via 160-entry table,
// so the A gather is ds_load_u16 with one add.  D store = one b128 per lane.
// c3o: f16 [B,16,12,12] (+bias)
// ---------------------------------------------------------------------------
__global__ void k3_c3_wmma(const _Float16* __restrict__ h2,
                           const _Float16* __restrict__ Wc3T,
                           const float* __restrict__ c3b,
                           _Float16* __restrict__ c3o) {
    __shared__ __align__(16) _Float16 h2s[1792];   // 1536 image + 256 zero pad
    __shared__ unsigned short ctab[160];
    int t = threadIdx.x;
    int b = blockIdx.x;
    if (t < 192) {
        ((uint4*)h2s)[t] = ((const uint4*)(h2 + (size_t)b * 1536))[t];
    } else if (t < 224) {                          // zero halves [1536,1792)
        uint4 z; z.x = z.y = z.z = z.w = 0u;
        ((uint4*)h2s)[t] = z;
    }
    if (t < 160) {
        int k = t;
        unsigned short off = 1536;
        if (k < 150) {
            int ci = k / 25, o = k % 25;
            off = (unsigned short)(ci * 256 + (o / 5) * 16 + (o % 5));
        }
        ctab[k] = off;
    }
    __syncthreads();

    int lane = t & 31, wv = t >> 5;                // wv in [0,9)
    int half = lane >> 4, m = lane & 15, n = lane & 15;
    int r = wv * 16 + m;                           // pixel id in image [0,144)
    int posoff = (r / 12) * 16 + (r % 12);         // oy*16 + ox
    const _Float16* wb = Wc3T + n * 160;

    v8f acc = {};
    #pragma unroll
    for (int kc = 0; kc < 160; kc += 32) {
        v16h a;
        #pragma unroll
        for (int e = 0; e < 8; ++e) {
            int k1 = kc + half * 8 + e;            // elems 0-7
            int k2 = kc + 16 + half * 8 + e;       // elems 8-15
            a[e]     = h2s[(int)ctab[k1] + posoff];
            a[e + 8] = h2s[(int)ctab[k2] + posoff];
        }
        v16h bf = frag_cat(*(const v8h*)(wb + kc + half * 8),
                           *(const v8h*)(wb + kc + 16 + half * 8));
        acc = __builtin_amdgcn_wmma_f32_16x16x32_f16(false, a, false, bf,
                                                     (short)0, acc, false, false);
    }
    float bias = c3b[n];
    v8h outv;
    #pragma unroll
    for (int r2 = 0; r2 < 8; ++r2) outv[r2] = (_Float16)(acc[r2] + bias);
    // rows r2 are 8 consecutive pixels -> single 16B store
    *(v8h*)(c3o + (size_t)b * 2304 + n * 144 + wv * 16 + half * 8) = outv;
}

// ---------------------------------------------------------------------------
// K4: S4 (2x2 avg pool + affine) + tanh, scattered directly into the C5
// im2col matrix A5[32768][416] (each pooled pixel feeds <=4 patch slots).
// ---------------------------------------------------------------------------
__global__ void k4_s4_im2col(const _Float16* __restrict__ c3o,
                             const float* __restrict__ s4w,
                             const float* __restrict__ s4b,
                             _Float16* __restrict__ A5) {
    int idx = blockIdx.x * 256 + threadIdx.x;      // < 8192*16*36 exactly
    int b = idx / 576, rem = idx % 576, c = rem / 36, q = rem % 36;
    int y = q / 6, x = q % 6;
    const _Float16* src = c3o + (size_t)b * 2304 + c * 144;
    float s = (float)src[(2 * y) * 12 + 2 * x]
            + (float)src[(2 * y) * 12 + 2 * x + 1]
            + (float)src[(2 * y + 1) * 12 + 2 * x]
            + (float)src[(2 * y + 1) * 12 + 2 * x + 1];
    _Float16 hv = (_Float16)squash((0.25f * s) * s4w[c] + s4b[c]);
    #pragma unroll
    for (int oy = 0; oy < 2; ++oy) {
        int ky = y - oy;
        if (ky < 0 || ky > 4) continue;
        #pragma unroll
        for (int ox = 0; ox < 2; ++ox) {
            int kx = x - ox;
            if (kx < 0 || kx > 4) continue;
            int p = (b << 2) + (oy << 1) + ox;
            A5[(size_t)p * 416 + c * 25 + ky * 5 + kx] = hv;
        }
    }
}

// ---------------------------------------------------------------------------
// K5: C5 as WMMA GEMM (dominant stage, ~3.15 GFLOP).
// M = 32768 positions, N = 128 (120 valid), K = 416 (13 WMMA steps).
// Block = 8 waves = 8 N-tiles of one M-tile.  A tile staged in LDS
// (row stride 424 halves, bank-conflict-free), fragments = ds_load_b128 x2;
// B fragments = global_load_b128 x2 from Wc5T.  Output written K-padded
// into A6[32768][128] in reshape(-1,120) row order (+bias), 2x b64 stores.
// ---------------------------------------------------------------------------
__global__ void k5_c5_wmma(const _Float16* __restrict__ A5,
                           const _Float16* __restrict__ Wc5T,
                           const float* __restrict__ c5b,
                           _Float16* __restrict__ A6) {
    __shared__ __align__(16) _Float16 As[16 * 424];
    int t = threadIdx.x;
    int mtile = blockIdx.x;                        // [0, 2048)
    int p0 = mtile * 16;
    for (int i = t; i < 832; i += 256) {           // 16 rows x 52 uint4
        int row = i / 52, c = i % 52;
        ((uint4*)As)[row * 53 + c] = ((const uint4*)A5)[(size_t)(p0 + row) * 52 + c];
    }
    __syncthreads();

    int lane = t & 31, ntile = t >> 5;
    int half = lane >> 4, m = lane & 15;
    int ncol = ntile * 16 + (lane & 15);
    const _Float16* ab = As + m * 424;
    const _Float16* wb = Wc5T + (size_t)ncol * 416;

    v8f acc = {};
    for (int kc = 0; kc < 416; kc += 32) {
        v16h a = frag_cat(*(const v8h*)(ab + kc + half * 8),
                          *(const v8h*)(ab + kc + 16 + half * 8));
        v16h bf = frag_cat(*(const v8h*)(wb + kc + half * 8),
                           *(const v8h*)(wb + kc + 16 + half * 8));
        acc = __builtin_amdgcn_wmma_f32_16x16x32_f16(false, a, false, bf,
                                                     (short)0, acc, false, false);
    }
    if (ncol < 120) {
        float bias = c5b[ncol];
        int o0 = ncol * 4;                         // flat offset within image
        int u = o0 / 120, j0 = o0 % 120;           // F6 row part / col
        int pp0 = mtile * 16 + half * 8;           // first of this lane's 8 rows
        #pragma unroll
        for (int g = 0; g < 2; ++g) {              // two groups of 4 positions
            int bb = (pp0 >> 2) + g;               // image id
            v4h outv;
            #pragma unroll
            for (int qq = 0; qq < 4; ++qq)
                outv[qq] = (_Float16)(acc[g * 4 + qq] + bias);
            *(v4h*)(A6 + (size_t)(bb * 4 + u) * 128 + j0) = outv;
        }
    }
}

// ---------------------------------------------------------------------------
// K6: F6 as WMMA GEMM.  M = 32768 reshaped rows, N = 96 (84 valid), K = 128.
// A rows are K-padded -> guard-free 2x global_load_b128; +bias, tanh.
// h6: f32 [32768,84]
// ---------------------------------------------------------------------------
__global__ void k6_f6_wmma(const _Float16* __restrict__ A6,
                           const _Float16* __restrict__ Wf6T,
                           const float* __restrict__ f6b,
                           float* __restrict__ h6) {
    int t = threadIdx.x;
    int lane = t & 31, wv = t >> 5;
    int gw = blockIdx.x * 8 + wv;                  // [0, 12288) = 2048 m x 6 n
    int mtile = gw / 6, ntile = gw % 6;
    int half = lane >> 4, m = lane & 15;
    int ncol = ntile * 16 + (lane & 15);
    const _Float16* ab = A6 + (size_t)(mtile * 16 + m) * 128;
    const _Float16* wb = Wf6T + (size_t)ncol * 128;

    v8f acc = {};
    #pragma unroll
    for (int kc = 0; kc < 128; kc += 32) {
        v16h a  = frag_cat(*(const v8h*)(ab + kc + half * 8),
                           *(const v8h*)(ab + kc + 16 + half * 8));
        v16h bf = frag_cat(*(const v8h*)(wb + kc + half * 8),
                           *(const v8h*)(wb + kc + 16 + half * 8));
        acc = __builtin_amdgcn_wmma_f32_16x16x32_f16(false, a, false, bf,
                                                     (short)0, acc, false, false);
    }
    if (ncol < 84) {
        float bias = f6b[ncol];
        #pragma unroll
        for (int r2 = 0; r2 < 8; ++r2) {
            int rr = mtile * 16 + r2 + half * 8;
            h6[(size_t)rr * 84 + ncol] = squash(acc[r2] + bias);
        }
    }
}

// ---------------------------------------------------------------------------
// K7: RBF layer: out[r,k] = sum_j (h6[r,j] - rbf_w[j,k])^2.  Prototypes in LDS.
// ---------------------------------------------------------------------------
__global__ void k7_rbf(const float* __restrict__ h6,
                       const float* __restrict__ rbfw,
                       float* __restrict__ out) {
    __shared__ float sw[840];
    int t = threadIdx.x;
    for (int i = t; i < 840; i += 256) sw[i] = rbfw[i];
    __syncthreads();
    int row = blockIdx.x * 256 + t;                // < 32768 exactly
    float acc[10];
    #pragma unroll
    for (int k = 0; k < 10; ++k) acc[k] = 0.f;
    const float* hr = h6 + (size_t)row * 84;
    for (int j = 0; j < 84; ++j) {
        float h = hr[j];
        #pragma unroll
        for (int k = 0; k < 10; ++k) {
            float d = h - sw[j * 10 + k];
            acc[k] = fmaf(d, d, acc[k]);
        }
    }
    #pragma unroll
    for (int k = 0; k < 10; ++k) out[(size_t)row * 10 + k] = acc[k];
}

extern "C" void kernel_launch(void* const* d_in, const int* in_sizes, int n_in,
                              void* d_out, int out_size, void* d_ws, size_t ws_size,
                              hipStream_t stream) {
    (void)in_sizes; (void)n_in; (void)out_size; (void)ws_size;
    const float* x    = (const float*)d_in[0];
    const float* c1w  = (const float*)d_in[1];
    const float* c1b  = (const float*)d_in[2];
    const float* s2w  = (const float*)d_in[3];
    const float* s2b  = (const float*)d_in[4];
    const float* c3w  = (const float*)d_in[5];
    const float* c3b  = (const float*)d_in[6];
    const float* s4w  = (const float*)d_in[7];
    const float* s4b  = (const float*)d_in[8];
    const float* c5w  = (const float*)d_in[9];
    const float* c5b  = (const float*)d_in[10];
    const float* f6w  = (const float*)d_in[11];
    const float* f6b  = (const float*)d_in[12];
    const float* rbfw = (const float*)d_in[13];

    // Workspace layout (256B-aligned, total ~104.6 MB -> L2-resident)
    char* ws = (char*)d_ws;
    _Float16* h2   = (_Float16*)(ws + 0ull);           // 8192*1536 f16 = 25165824 B
    _Float16* c3o  = (_Float16*)(ws + 25165824ull);    // 8192*2304 f16 = 37748736 B
    _Float16* A5   = (_Float16*)(ws + 62914560ull);    // 32768*416 f16 = 27262976 B
    _Float16* A6   = (_Float16*)(ws + 90177536ull);    // 32768*128 f16 =  8388608 B
    float*    h6   = (float*)   (ws + 98566144ull);    // 32768*84  f32 = 11010048 B
    _Float16* Wc3T = (_Float16*)(ws + 109576192ull);   // 16*160  f16
    _Float16* Wc5T = (_Float16*)(ws + 109581312ull);   // 128*416 f16
    _Float16* Wf6T = (_Float16*)(ws + 109687808ull);   // 96*128  f16

    k1_c1_s2     <<<8192, 256, 0, stream>>>(x, c1w, c1b, s2w, s2b, h2);
    k2_pack      <<<3338, 256, 0, stream>>>(c3w, c5w, f6w, Wc3T, Wc5T, Wf6T, A5, A6);
    k3_c3_wmma   <<<8192, 288, 0, stream>>>(h2, Wc3T, c3b, c3o);
    k4_s4_im2col <<<18432, 256, 0, stream>>>(c3o, s4w, s4b, A5);
    k5_c5_wmma   <<<2048, 256, 0, stream>>>(A5, Wc5T, c5b, A6);
    k6_f6_wmma   <<<1536, 256, 0, stream>>>(A6, Wf6T, f6b, h6);
    k7_rbf       <<<128, 256, 0, stream>>>(h6, rbfw, (float*)d_out);
}
